// GraphLaplacianAttention_72009421685335
// MI455X (gfx1250) — matrix operand order repaired and checked by reading, not compiled, not run on hardware
//
#include <hip/hip_runtime.h>
#include <hip/hip_bf16.h>
#include <stdint.h>

// ---------------------------------------------------------------------------
// GraphLaplacianAttention on MI455X (gfx1250, wave32, WMMA 16x16x32 f16)
// Fragment-major LDS staging -> ds_load_b128 operand loads (no u16 gathers).
// ---------------------------------------------------------------------------

#define TDIM 128      // feature dim (= EH*HD = H*HD_q)
#define NH   8        // heads
#define NEH  8        // expanded heads
#define HDQ  16       // head dim

typedef _Float16 v16h __attribute__((ext_vector_type(16)));
typedef _Float16 v8h  __attribute__((ext_vector_type(8)));
typedef float    v8f  __attribute__((ext_vector_type(8)));

// sAf: 16 frags (4 waves x 4 kb)  * 32 lanes * 16 halves = 8192 halves (16KB)
// sBf: 32 frags (4 kb x 8 nt)     * 32 lanes * 16 halves = 16384 halves (32KB)
#define AFRAG_HALVES (16 * 32 * 16)
#define BFRAG_HALVES (32 * 32 * 16)

// order-preserving encoding of f32 into u32 so unsigned atomicMax == float max
__device__ __forceinline__ unsigned encodeF(float x) {
  unsigned u = __float_as_uint(x);
  return (u & 0x80000000u) ? ~u : (u | 0x80000000u);
}
__device__ __forceinline__ float decodeF(unsigned e) {
  return (e & 0x80000000u) ? __uint_as_float(e & 0x7FFFFFFFu)
                           : __uint_as_float(~e);
}

// ---------------------------------------------------------------------------
// Stage 64 rows of a row-major f32 matrix (row stride TDIM) into A-fragment
// layout.  A 16-bit 16x32 fragment (ISA 7.12.2): lane = half*16 + (row&15),
// halves idx 0..7 <-> K = k0..k0+7, idx 8..15 <-> K = k0+16..k0+23, where
// half = (Klocal>>3)&1.  Each thread: 8 coalesced f32 reads -> 1 b128 store.
// ---------------------------------------------------------------------------
__device__ __forceinline__ void stage_A_rows(const float* __restrict__ g,
                                             int r0, int rMax,
                                             _Float16* __restrict__ sAf) {
  for (int task = threadIdx.x; task < 1024; task += 128) {
    const int row = task >> 4;          // 0..63
    const int k0  = (task & 15) << 3;   // 0..120 step 8 (consecutive per lane)
    int gr = r0 + row; if (gr >= rMax) gr = rMax - 1;
    const float* src = g + (size_t)gr * TDIM + k0;
    v8h h;
#pragma unroll
    for (int j = 0; j < 8; ++j) h[j] = (_Float16)src[j];
    const int wave = row >> 4;
    const int kb   = k0 >> 5;
    const int kl   = k0 & 31;
    const int half = (kl >> 3) & 1;
    const int lane = (half << 4) + (row & 15);
    const int idx0 = (kl & 16) >> 1;    // 0 or 8
    *(v8h*)&sAf[(((wave << 2) + kb) * 32 + lane) * 16 + idx0] = h;
  }
}

// Same but A := u - w (elementwise), for the output projection.
__device__ __forceinline__ void stage_A_rows_diff(const float* __restrict__ u,
                                                  const float* __restrict__ w,
                                                  int r0, int rMax,
                                                  _Float16* __restrict__ sAf) {
  for (int task = threadIdx.x; task < 1024; task += 128) {
    const int row = task >> 4;
    const int k0  = (task & 15) << 3;
    int gr = r0 + row; if (gr >= rMax) gr = rMax - 1;
    const size_t base = (size_t)gr * TDIM + k0;
    v8h h;
#pragma unroll
    for (int j = 0; j < 8; ++j) h[j] = (_Float16)(u[base + j] - w[base + j]);
    const int wave = row >> 4;
    const int kb   = k0 >> 5;
    const int kl   = k0 & 31;
    const int half = (kl >> 3) & 1;
    const int lane = (half << 4) + (row & 15);
    const int idx0 = (kl & 16) >> 1;
    *(v8h*)&sAf[(((wave << 2) + kb) * 32 + lane) * 16 + idx0] = h;
  }
}

// ---------------------------------------------------------------------------
// Stage a 128x128 row-major f32 weight into B-fragment layout.
// B 16-bit 32x16 fragment: lane = half*16 + (col&15), half = (Klocal>>3)&1,
// halves idx 0..7 <-> K = k0..k0+7, idx 8..15 <-> K = k0+16..k0+23.  8 coalesced reads -> 1 b128.
// ---------------------------------------------------------------------------
__device__ __forceinline__ void stage_B_weight(const float* __restrict__ W,
                                               _Float16* __restrict__ sBf) {
  for (int task = threadIdx.x; task < 2048; task += 128) {
    const int col = task & 127;
    const int k0  = (task >> 7) << 3;   // 0..120 step 8
    const float* src = W + (size_t)k0 * TDIM + col;
    v8h h;
#pragma unroll
    for (int j = 0; j < 8; ++j) h[j] = (_Float16)src[(size_t)j * TDIM];
    const int kb   = k0 >> 5;
    const int half = (k0 >> 3) & 1;
    const int nt   = col >> 4;
    const int lane = (half << 4) + (col & 15);
    const int idx0 = (k0 & 16) >> 1;    // 0 or 8
    *(v8h*)&sBf[(((kb << 3) + nt) * 32 + lane) * 16 + idx0] = h;
  }
}

// ---------------------------------------------------------------------------
// Core: 64x128 += A(64x128 f16) @ B(128x128 f16), fragment-major LDS.
// Wave w owns rows [16w,16w+16); one v16h load per operand (2x ds_load_b128).
// ---------------------------------------------------------------------------
__device__ __forceinline__ void wmma_gemm_64x128(const _Float16* __restrict__ sAf,
                                                 const _Float16* __restrict__ sBf,
                                                 v8f acc[8]) {
  const int lane = threadIdx.x & 31;
  const int wave = threadIdx.x >> 5;
#pragma unroll
  for (int kb = 0; kb < 4; ++kb) {
    const v16h a = *(const v16h*)&sAf[(((wave << 2) + kb) * 32 + lane) * 16];
#pragma unroll
    for (int nt = 0; nt < 8; ++nt) {
      const v16h b = *(const v16h*)&sBf[(((kb << 3) + nt) * 32 + lane) * 16];
      acc[nt] = __builtin_amdgcn_wmma_f32_16x16x32_f16(
          false, a, false, b, (short)0, acc[nt], false, false);
    }
  }
}

__device__ __forceinline__ void zero_acc(v8f acc[8]) {
#pragma unroll
  for (int t = 0; t < 8; ++t)
#pragma unroll
    for (int j = 0; j < 8; ++j) acc[t][j] = 0.0f;
}

// C/D layout: lane holds col (lane&15), rows m + 8*(lane>>4) for VGPR m.
__device__ __forceinline__ void store_C_global(const v8f acc[8],
                                               float* __restrict__ out,
                                               int r0, int rMax) {
  const int lane = threadIdx.x & 31, wave = threadIdx.x >> 5;
  const int l15 = lane & 15, mOff = (lane >> 4) * 8;
#pragma unroll
  for (int nt = 0; nt < 8; ++nt)
#pragma unroll
    for (int m = 0; m < 8; ++m) {
      int gr = r0 + wave * 16 + mOff + m;
      if (gr < rMax) out[(size_t)gr * TDIM + (nt << 4) + l15] = acc[nt][m];
    }
}

// Write C tile into LDS row-major (rows of 128 halves, 32B-aligned).
__device__ __forceinline__ void store_C_lds_f16(const v8f acc[8],
                                                _Float16* __restrict__ sT) {
  const int lane = threadIdx.x & 31, wave = threadIdx.x >> 5;
  const int l15 = lane & 15, mOff = (lane >> 4) * 8;
#pragma unroll
  for (int nt = 0; nt < 8; ++nt)
#pragma unroll
    for (int m = 0; m < 8; ++m)
      sT[(wave * 16 + mOff + m) * TDIM + (nt << 4) + l15] = (_Float16)acc[nt][m];
}

// ---------------------------------------------------------------------------
// init: mx (encoded -inf == 0), denom = 0, agg = 0
// ---------------------------------------------------------------------------
__global__ void gla_init_kernel(unsigned* __restrict__ mx,
                                float* __restrict__ denom,
                                float* __restrict__ agg, int n) {
  int idx = blockIdx.x * blockDim.x + threadIdx.x;
  if (idx < n * NEH) { mx[idx] = 0u; denom[idx] = 0.0f; }
  if (idx < n * TDIM) agg[idx] = 0.0f;
}

// ---------------------------------------------------------------------------
// node projections: q = x@Wq, k = x@Wk, v = x@Wv   (blockIdx.z selects)
// ---------------------------------------------------------------------------
__global__ __launch_bounds__(128) void gla_node_proj_kernel(
    const float* __restrict__ x, const float* __restrict__ Wq,
    const float* __restrict__ Wk, const float* __restrict__ Wv,
    float* __restrict__ q, float* __restrict__ k, float* __restrict__ v,
    int n) {
  __shared__ __align__(32) _Float16 sAf[AFRAG_HALVES];
  __shared__ __align__(32) _Float16 sBf[BFRAG_HALVES];
  const float* W = (blockIdx.z == 0) ? Wq : (blockIdx.z == 1) ? Wk : Wv;
  float* out     = (blockIdx.z == 0) ? q  : (blockIdx.z == 1) ? k  : v;
  const int r0 = blockIdx.x * 64;

  stage_B_weight(W, sBf);
  stage_A_rows(x, r0, n, sAf);
  __syncthreads();

  v8f acc[8]; zero_acc(acc);
  wmma_gemm_64x128(sAf, sBf, acc);
  store_C_global(acc, out, r0, n);
}

// ---------------------------------------------------------------------------
// edge pass 1: ek = edges@Wek (WMMA); attn_h = scale * q[src]·(k[dst]+ek);
// attnEH = attn @ W_exp; segment max via encoded atomicMax.
// ---------------------------------------------------------------------------
__global__ __launch_bounds__(128) void gla_edge_attn_kernel(
    const float* __restrict__ edges, const float* __restrict__ Wek,
    const float* __restrict__ q, const float* __restrict__ k,
    const float* __restrict__ Wexp, const int* __restrict__ srcIdx,
    const int* __restrict__ dstIdx, float* __restrict__ attnEH,
    unsigned* __restrict__ mx, int nE) {
  __shared__ __align__(32) _Float16 sAf[AFRAG_HALVES];  // edge frags, then ek
  __shared__ __align__(32) _Float16 sBf[BFRAG_HALVES];  // Wek frags
  __shared__ float sAttn[64 * NH];
  __shared__ float sWexp[NH * NEH];
  __shared__ int sSrc[64], sDst[64];
  const int e0 = blockIdx.x * 64;

  stage_B_weight(Wek, sBf);
  stage_A_rows(edges, e0, nE, sAf);
  if (threadIdx.x < 64) {
    int ge = e0 + threadIdx.x; if (ge >= nE) ge = nE - 1;
    sSrc[threadIdx.x] = srcIdx[ge];
    sDst[threadIdx.x] = dstIdx[ge];
    sWexp[threadIdx.x] = Wexp[threadIdx.x];
  }
  __syncthreads();

  v8f acc[8]; zero_acc(acc);
  wmma_gemm_64x128(sAf, sBf, acc);
  __syncthreads();                       // all waves done reading edge frags

  _Float16* sEk = sAf;                   // reuse as row-major 64x128 ek tile
  store_C_lds_f16(acc, sEk);
  __syncthreads();

  const float scale = 0.25f;             // HD^-0.5
#pragma unroll
  for (int i = 0; i < 4; ++i) {          // 512 (edge,head) dot products
    int f = i * 128 + threadIdx.x;
    int e = f >> 3, h = f & 7;
    const float4* qr = (const float4*)(q + (size_t)sSrc[e] * TDIM + h * HDQ);
    const float4* kr = (const float4*)(k + (size_t)sDst[e] * TDIM + h * HDQ);
    const v16h ek = *(const v16h*)&sEk[e * TDIM + h * HDQ];
    float s = 0.0f;
#pragma unroll
    for (int t4 = 0; t4 < 4; ++t4) {
      float4 qv = qr[t4], kv = kr[t4];
      s += qv.x * (kv.x + (float)ek[4 * t4 + 0]);
      s += qv.y * (kv.y + (float)ek[4 * t4 + 1]);
      s += qv.z * (kv.z + (float)ek[4 * t4 + 2]);
      s += qv.w * (kv.w + (float)ek[4 * t4 + 3]);
    }
    sAttn[e * NH + h] = s * scale;
  }
  __syncthreads();

#pragma unroll
  for (int i = 0; i < 4; ++i) {          // attn @ W_exp (8x8) + segment max
    int f = i * 128 + threadIdx.x;
    int e = f >> 3, j = f & 7;
    float val = 0.0f;
#pragma unroll
    for (int h = 0; h < NH; ++h) val += sAttn[e * NH + h] * sWexp[h * NEH + j];
    int ge = e0 + e;
    if (ge < nE) {
      attnEH[(size_t)ge * NEH + j] = val;
      atomicMax(&mx[(size_t)sSrc[e] * NEH + j], encodeF(val));
    }
  }
}

// ---------------------------------------------------------------------------
// denom: atomicAdd of exp(attn - mx[src])
// ---------------------------------------------------------------------------
__global__ void gla_denom_kernel(const float* __restrict__ attnEH,
                                 const int* __restrict__ srcIdx,
                                 const unsigned* __restrict__ mx,
                                 float* __restrict__ denom, int nE) {
  int idx = blockIdx.x * blockDim.x + threadIdx.x;
  if (idx >= nE * NEH) return;
  int e = idx >> 3, j = idx & 7;
  int s = srcIdx[e];
  float ex = __expf(attnEH[idx] - decodeF(mx[(size_t)s * NEH + j]));
  atomicAdd(&denom[(size_t)s * NEH + j], ex);
}

// ---------------------------------------------------------------------------
// edge pass 2: ev = edges@Wev (WMMA, recomputed to avoid 410MB spill);
// coef = exp(attn-mx)/denom; agg[src] += coef*(v[dst]+ev)  (coalesced atomics)
// ---------------------------------------------------------------------------
__global__ __launch_bounds__(128) void gla_edge_agg_kernel(
    const float* __restrict__ edges, const float* __restrict__ Wev,
    const float* __restrict__ v, const float* __restrict__ attnEH,
    const unsigned* __restrict__ mx, const float* __restrict__ denom,
    const int* __restrict__ srcIdx, const int* __restrict__ dstIdx,
    float* __restrict__ agg, int nE) {
  __shared__ __align__(32) _Float16 sAf[AFRAG_HALVES];  // edge frags, then ev
  __shared__ __align__(32) _Float16 sBf[BFRAG_HALVES];  // Wev frags
  __shared__ float sCoef[64 * NEH];
  __shared__ int sSrc[64], sDst[64];
  const int e0 = blockIdx.x * 64;

  stage_B_weight(Wev, sBf);
  stage_A_rows(edges, e0, nE, sAf);
  if (threadIdx.x < 64) {
    int ge = e0 + threadIdx.x; if (ge >= nE) ge = nE - 1;
    sSrc[threadIdx.x] = srcIdx[ge];
    sDst[threadIdx.x] = dstIdx[ge];
  }
  __syncthreads();

#pragma unroll
  for (int i = 0; i < 4; ++i) {          // softmax coefficients
    int f = i * 128 + threadIdx.x;
    int e = f >> 3, j = f & 7;
    int ge = e0 + e; if (ge >= nE) ge = nE - 1;
    int s = sSrc[e];
    float a = __expf(attnEH[(size_t)ge * NEH + j] -
                     decodeF(mx[(size_t)s * NEH + j])) /
              denom[(size_t)s * NEH + j];
    sCoef[e * NEH + j] = a;
  }

  v8f acc[8]; zero_acc(acc);
  wmma_gemm_64x128(sAf, sBf, acc);
  __syncthreads();

  _Float16* sEv = sAf;                   // reuse as row-major 64x128 ev tile
  store_C_lds_f16(acc, sEv);
  __syncthreads();

  const int d = threadIdx.x;             // block handles one edge-row/iter
  for (int e = 0; e < 64; ++e) {
    int ge = e0 + e;
    if (ge >= nE) break;
    float val = sCoef[e * NEH + (d >> 4)] *
                (v[(size_t)sDst[e] * TDIM + d] + (float)sEv[e * TDIM + d]);
    atomicAdd(&agg[(size_t)sSrc[e] * TDIM + d], val);
  }
}

// ---------------------------------------------------------------------------
// output: out = (v - agg) @ W_out
// ---------------------------------------------------------------------------
__global__ __launch_bounds__(128) void gla_out_kernel(
    const float* __restrict__ v, const float* __restrict__ agg,
    const float* __restrict__ Wout, float* __restrict__ out, int n) {
  __shared__ __align__(32) _Float16 sAf[AFRAG_HALVES];
  __shared__ __align__(32) _Float16 sBf[BFRAG_HALVES];
  const int r0 = blockIdx.x * 64;

  stage_B_weight(Wout, sBf);
  stage_A_rows_diff(v, agg, r0, n, sAf);
  __syncthreads();

  v8f acc[8]; zero_acc(acc);
  wmma_gemm_64x128(sAf, sBf, acc);
  store_C_global(acc, out, r0, n);
}

// ---------------------------------------------------------------------------
extern "C" void kernel_launch(void* const* d_in, const int* in_sizes, int n_in,
                              void* d_out, int out_size, void* d_ws,
                              size_t ws_size, hipStream_t stream) {
  const float* x     = (const float*)d_in[0];
  const float* edges = (const float*)d_in[1];
  const float* Wq    = (const float*)d_in[2];
  const float* Wk    = (const float*)d_in[3];
  const float* Wv    = (const float*)d_in[4];
  const float* Wek   = (const float*)d_in[5];
  const float* Wev   = (const float*)d_in[6];
  const float* Wexp  = (const float*)d_in[7];
  const float* Wout  = (const float*)d_in[8];
  const int*   eidx  = (const int*)d_in[9];   // int64 in source, int32 on device (jax x64 off)

  const int n = in_sizes[0] / TDIM;
  const int e = in_sizes[1] / TDIM;
  const int* srcIdx = eidx;
  const int* dstIdx = eidx + e;

  char* ws = (char*)d_ws;
  size_t off = 0;
  auto alloc = [&](size_t bytes) -> void* {
    void* p = ws + off;
    off = (off + bytes + 255) & ~(size_t)255;
    return p;
  };
  float*    q      = (float*)alloc((size_t)n * TDIM * sizeof(float));
  float*    kbuf   = (float*)alloc((size_t)n * TDIM * sizeof(float));
  float*    vbuf   = (float*)alloc((size_t)n * TDIM * sizeof(float));
  float*    agg    = (float*)alloc((size_t)n * TDIM * sizeof(float));
  float*    attnEH = (float*)alloc((size_t)e * NEH * sizeof(float));
  unsigned* mx     = (unsigned*)alloc((size_t)n * NEH * sizeof(unsigned));
  float*    denom  = (float*)alloc((size_t)n * NEH * sizeof(float));
  (void)ws_size; (void)n_in; (void)out_size;

  // 1) init accumulation buffers
  {
    int total = n * TDIM;
    gla_init_kernel<<<(total + 255) / 256, 256, 0, stream>>>(mx, denom, agg, n);
  }
  // 2) node projections (z picks Wq/Wk/Wv)
  {
    dim3 grid((n + 63) / 64, 1, 3);
    gla_node_proj_kernel<<<grid, 128, 0, stream>>>(x, Wq, Wk, Wv, q, kbuf, vbuf, n);
  }
  // 3) edge attention logits + segment max
  gla_edge_attn_kernel<<<(e + 63) / 64, 128, 0, stream>>>(
      edges, Wek, q, kbuf, Wexp, srcIdx, dstIdx, attnEH, mx, e);
  // 4) softmax denominator
  {
    int total = e * NEH;
    gla_denom_kernel<<<(total + 255) / 256, 256, 0, stream>>>(
        attnEH, srcIdx, mx, denom, e);
  }
  // 5) weighted aggregation into agg
  gla_edge_agg_kernel<<<(e + 63) / 64, 128, 0, stream>>>(
      edges, Wev, vbuf, attnEH, mx, denom, srcIdx, dstIdx, agg, e);
  // 6) output projection
  gla_out_kernel<<<(n + 63) / 64, 128, 0, stream>>>(
      vbuf, agg, Wout, (float*)d_out, n);
}
